// MultiHeadAttentionFusion_47631187313312
// MI455X (gfx1250) — matrix operand order repaired and checked by reading, not compiled, not run on hardware
//
#include <hip/hip_runtime.h>
#include <hip/hip_bf16.h>

typedef __attribute__((ext_vector_type(16))) __bf16 v16bf;
typedef __attribute__((ext_vector_type(8)))  float  v8f;
typedef __attribute__((ext_vector_type(4)))  unsigned int v4u;
typedef __attribute__((ext_vector_type(8)))  int    v8i_t;
typedef __attribute__((ext_vector_type(4)))  int    v4i_t;

#define WMMA_BF16(a,b,c) __builtin_amdgcn_wmma_f32_16x16x32_bf16(false,(a),false,(b),(short)0,(c),false,false)

// ---- CDNA5 async / TDM availability probes (compile-safe fallbacks) ----
#if __has_builtin(__builtin_amdgcn_global_load_async_to_lds_b128) && \
    __has_builtin(__builtin_amdgcn_s_wait_asynccnt)
#define HAVE_ASYNC 1
#else
#define HAVE_ASYNC 0
#endif

#if __has_builtin(__builtin_amdgcn_tensor_load_to_lds) && \
    __has_builtin(__builtin_amdgcn_s_wait_tensorcnt)
#define HAVE_TDM 1
#else
#define HAVE_TDM 0
#endif

#define AS1 __attribute__((address_space(1)))
#define AS3 __attribute__((address_space(3)))

#if HAVE_ASYNC
// 16-byte async copy global -> LDS (ASYNCcnt-tracked, no VGPR data round trip)
// Builtin signature (from clang diagnostic): (AS1 v4i*, AS3 v4i*, imm offset, imm cpol)
__device__ __forceinline__ void async_b128(const void* g, void* l) {
  __builtin_amdgcn_global_load_async_to_lds_b128(
      (AS1 v4i_t*)(uintptr_t)g,
      (AS3 v4i_t*)(uint32_t)(uintptr_t)l, 0, 0);
}
#endif

constexpr int BB = 8, S = 1024, D = 512, H = 8, DK = 64, DV = 64;
constexpr int ROWS = BB * S; // 8192

// ---- WMMA fragment gathers from LDS, contiguous k order, k = 16*(lane/16) + e ----
__device__ __forceinline__ v16bf frag_a(const __bf16* t, int ld) {
  int lane = threadIdx.x & 31;
  int m = lane & 15;
  int kb = (lane >> 4) << 4;
  v16bf a;
#pragma unroll
  for (int e = 0; e < 16; ++e) {
    int v = e >> 1, p = e & 1;
    int k = kb + 2 * v + p;
    a[e] = t[m * ld + k];
  }
  return a;
}
__device__ __forceinline__ v16bf frag_b(const __bf16* t, int ld, int n0) {
  int lane = threadIdx.x & 31;
  int n = lane & 15;
  int kb = (lane >> 4) << 4;
  v16bf b;
#pragma unroll
  for (int e = 0; e < 16; ++e) {
    int v = e >> 1, p = e & 1;
    b[e] = t[(kb + 2 * v + p) * ld + n0 + n];
  }
  return b;
}
__device__ __forceinline__ v16bf frag_bT(const __bf16* t, int ld, int n0, int kk) {
  int lane = threadIdx.x & 31;
  int n = lane & 15;
  int kb = (lane >> 4) << 4;
  v16bf b;
#pragma unroll
  for (int e = 0; e < 16; ++e) {
    int v = e >> 1, p = e & 1;
    b[e] = t[(n0 + n) * ld + kk + kb + 2 * v + p];
  }
  return b;
}

// ================= Kernel 1: fused Q/K/V projections (bf16 out) ===========
__global__ __launch_bounds__(256) void proj_kernel(
    const float* __restrict__ q, const float* __restrict__ k,
    const float* __restrict__ vq, const float* __restrict__ vk,
    const float* __restrict__ Wq, const float* __restrict__ Wk,
    const float* __restrict__ Wv,
    __bf16* __restrict__ Qh, __bf16* __restrict__ Kh, __bf16* __restrict__ Vh) {
  constexpr int LDA = 34, LDB = 68;
  __shared__ __align__(16) __bf16 Asm[64 * LDA];
  __shared__ __align__(16) __bf16 Bsm[32 * LDB];
  const int tid = threadIdx.x, lane = tid & 31, wid = tid >> 5;
  const int which = blockIdx.z;
  const int rowBase = blockIdx.y * 64, colBase = blockIdx.x * 64;
  const int wm = wid & 1, wn = wid >> 1;
  const float* X0 = (which == 0) ? q : (which == 1) ? k : vq;
  const float* W  = (which == 0) ? Wq : (which == 1) ? Wk : Wv;
  __bf16* dst = (which == 0) ? Qh : (which == 1) ? Kh : Vh;
  const float scale = (which == 0) ? 0.125f : 1.0f;
  v8f acc0 = {}, acc1 = {};
  for (int kk = 0; kk < D; kk += 32) {
    for (int i = tid; i < 64 * 32; i += 256) {
      int r = i >> 5, c = i & 31;
      float x = X0[(rowBase + r) * D + kk + c];
      if (which == 2) x += vk[(rowBase + r) * D + kk + c];
      Asm[r * LDA + c] = (__bf16)x;
    }
    for (int i = tid; i < 32 * 64; i += 256) {
      int r = i >> 6, c = i & 63;
      Bsm[r * LDB + c] = (__bf16)W[(kk + r) * D + colBase + c];
    }
    __syncthreads();
    v16bf a0 = frag_a(&Asm[(wm * 32) * LDA], LDA);
    v16bf a1 = frag_a(&Asm[(wm * 32 + 16) * LDA], LDA);
    v16bf bb = frag_b(Bsm, LDB, wn * 16);
    acc0 = WMMA_BF16(a0, bb, acc0);
    acc1 = WMMA_BF16(a1, bb, acc1);
    __syncthreads();
  }
  int n = lane & 15, mb = (lane >> 4) << 3;
  int col = colBase + wn * 16 + n;
#pragma unroll
  for (int r = 0; r < 8; ++r) {
    dst[(rowBase + wm * 32 + mb + r) * D + col]      = (__bf16)(acc0[r] * scale);
    dst[(rowBase + wm * 32 + 16 + mb + r) * D + col] = (__bf16)(acc1[r] * scale);
  }
}

// ================= Kernel 2: attn = Qh Kh^T, symmetric mask, store fp32 ====
__global__ __launch_bounds__(256) void scores_kernel(
    const __bf16* __restrict__ Qh, const __bf16* __restrict__ Kh,
    const int* __restrict__ mask, float* __restrict__ attn) {
  __shared__ __align__(16) __bf16 Qs[16 * 64];
  __shared__ __align__(16) __bf16 Ks[128 * 64];
  const int tid = threadIdx.x, lane = tid & 31, wid = tid >> 5;
  const int bh = blockIdx.z, b = bh >> 3, h = bh & 7;
  const int qbase = blockIdx.y * 16, kbase = blockIdx.x * 128;

  // ---- stage K tile: 128 x 64 bf16 (16 KB), stride-512 2D tile ----
#if HAVE_TDM
  if (wid == 0) {
    // Tensor DMA descriptor (D#): one 2D tile, data_size=2B,
    // tile_dim0=64 (x), tile_dim1=128 (y), tensor_dim0_stride=512 elems.
    uint64_t ga = (uint64_t)(uintptr_t)(Kh + (size_t)(b * S + kbase) * D + h * DK);
    uint32_t ldsoff = (uint32_t)(uintptr_t)(void*)&Ks[0];
    v4u g0 = { 1u,                                   // count=1, no gather
               ldsoff,                               // lds_addr
               (uint32_t)ga,                         // global_addr[31:0]
               (uint32_t)((ga >> 32) & 0x01FFFFFFu) | 0x80000000u }; // addr[56:32], type=2
    v8i_t g1 = { (int)0x00010000,   // wg_mask=0, data_size=1 (2 bytes)
                 (int)0x00400000,   // tensor_dim0=64 -> bits[79:48] low half
                 (int)0x00800000,   // tensor_dim1=128 -> bits[111:80] low half
                 (int)0x00400000,   // tile_dim0=64 -> bits[127:112]
                 (int)0x00000080,   // tile_dim1=128 -> bits[143:128]
                 512, 0, 0 };       // tensor_dim0_stride=512
    v4i_t z = { 0, 0, 0, 0 };
#if __clang_major__ >= 23
    v8i_t z8 = { 0, 0, 0, 0, 0, 0, 0, 0 };
    __builtin_amdgcn_tensor_load_to_lds(g0, g1, z, z, z8, 0);
#else
    __builtin_amdgcn_tensor_load_to_lds(g0, g1, z, z, 0);
#endif
    __builtin_amdgcn_s_wait_tensorcnt(0);
  }
#elif HAVE_ASYNC
  for (int i = tid; i < 1024; i += 256) { // 1024 x 16B chunks
    int r = i >> 3, c = i & 7;
    async_b128(Kh + (size_t)(b * S + kbase + r) * D + h * DK + c * 8,
               (char*)Ks + (r * 64 + c * 8) * 2);
  }
#else
  for (int i = tid; i < 128 * 32; i += 256) {
    int r = i >> 5, c = i & 31;
    ((uint32_t*)Ks)[i] =
        ((const uint32_t*)Kh)[(((b * S + kbase + r) * D + h * DK) >> 1) + c];
  }
#endif

  // ---- stage Q tile: 16 x 64 bf16 (2 KB) ----
#if HAVE_ASYNC
  if (tid < 128) { // 128 x 16B chunks
    int r = tid >> 3, c = tid & 7;
    async_b128(Qh + (size_t)(b * S + qbase + r) * D + h * DK + c * 8,
               (char*)Qs + (r * 64 + c * 8) * 2);
  }
  __builtin_amdgcn_s_wait_asynccnt(0);
#else
  for (int i = tid; i < 16 * 32; i += 256) {
    int r = i >> 5, c = i & 31;
    ((uint32_t*)Qs)[i] =
        ((const uint32_t*)Qh)[(((b * S + qbase + r) * D + h * DK) >> 1) + c];
  }
#endif
  __syncthreads();

  v8f acc = {};
#pragma unroll
  for (int kk = 0; kk < DK; kk += 32) {
    v16bf a = frag_a(&Qs[kk], 64);
    v16bf bb = frag_bT(Ks, 64, wid * 16, kk);
    acc = WMMA_BF16(a, bb, acc);
  }
  int n = lane & 15, mb = (lane >> 4) << 3;
  int kpos = kbase + wid * 16 + n;
#pragma unroll
  for (int r = 0; r < 8; ++r) {
    int qpos = qbase + mb + r;
    int m1 = mask[(b * S + qpos) * S + kpos];
    int m2 = mask[(b * S + kpos) * S + qpos];
    float val = (m1 == 0 || m2 == 0) ? -1.0e9f : acc[r];
    attn[((size_t)bh * S + qpos) * (size_t)S + kpos] = val;
  }
}

// ================= Kernel 3: out_h = attn @ Vh =============================
__global__ __launch_bounds__(128) void attnv_kernel(
    const float* __restrict__ attn, const __bf16* __restrict__ Vh,
    float* __restrict__ OutH) {
  __shared__ __align__(16) __bf16 As[16 * 32];
  __shared__ __align__(16) __bf16 Vs[32 * 64];
  const int tid = threadIdx.x, lane = tid & 31, wid = tid >> 5;
  const int bh = blockIdx.y, b = bh >> 3, h = bh & 7;
  const int qbase = blockIdx.x * 16;
  v8f acc = {};
  for (int kk = 0; kk < S; kk += 32) {
    // V tile: async copy first so it overlaps the fp32->bf16 conversion below
#if HAVE_ASYNC
    for (int i = tid; i < 256; i += 128) { // 256 x 16B chunks
      int r = i >> 3, c = i & 7;
      async_b128(Vh + (size_t)(b * S + kk + r) * D + h * DV + c * 8,
                 (char*)Vs + (r * 64 + c * 8) * 2);
    }
#else
    for (int i = tid; i < 32 * 32; i += 128) {
      int r = i >> 5, c = i & 31;
      ((uint32_t*)Vs)[i] =
          ((const uint32_t*)Vh)[(((b * S + kk + r) * D + h * DV) >> 1) + c];
    }
#endif
    for (int i = tid; i < 16 * 32; i += 128) {
      int r = i >> 5, c = i & 31;
      As[i] = (__bf16)attn[((size_t)bh * S + qbase + r) * (size_t)S + kk + c];
    }
#if HAVE_ASYNC
    __builtin_amdgcn_s_wait_asynccnt(0);
#endif
    __syncthreads();
    v16bf a = frag_a(As, 32);
    v16bf bb = frag_b(Vs, 64, wid * 16);
    acc = WMMA_BF16(a, bb, acc);
    __syncthreads();
  }
  int n = lane & 15, mb = (lane >> 4) << 3;
#pragma unroll
  for (int r = 0; r < 8; ++r)
    OutH[(b * S + qbase + mb + r) * D + h * DV + wid * 16 + n] = acc[r];
}

// ========== Kernel 4: (OutH @ Wfc) + 0.5*(vq+vk), LayerNorm, store =========
__global__ __launch_bounds__(256) void final_kernel(
    const float* __restrict__ OutH, const float* __restrict__ Wfc,
    const float* __restrict__ vq, const float* __restrict__ vk,
    const float* __restrict__ gamma, const float* __restrict__ beta,
    float* __restrict__ out) {
  constexpr int LDB = 520, LDA = 36;
  __shared__ __align__(16) float smemf[9000];
  __bf16* Bs = (__bf16*)smemf;
  __bf16* As = (__bf16*)(smemf + 8320);
  float* lnt   = smemf;
  float* red   = smemf + 8192;
  float* musig = smemf + 8704;
  const int tid = threadIdx.x, lane = tid & 31, wid = tid >> 5;
  const int rowBase = blockIdx.x * 16;
  v8f acc[4] = {};
  for (int kk = 0; kk < D; kk += 32) {
    for (int i = tid; i < 16 * 32; i += 256) {
      int r = i >> 5, c = i & 31;
      As[r * LDA + c] = (__bf16)OutH[(rowBase + r) * D + kk + c];
    }
    for (int i = tid; i < 32 * 512; i += 256) {
      int r = i >> 9, c = i & 511;
      Bs[r * LDB + c] = (__bf16)Wfc[(kk + r) * D + c];
    }
    if (kk + 32 < D) __builtin_prefetch(&Wfc[(kk + 32) * D], 0, 1);
    __syncthreads();
    v16bf a = frag_a(As, LDA);
#pragma unroll
    for (int t = 0; t < 4; ++t) {
      v16bf bb = frag_b(Bs, LDB, wid * 64 + t * 16);
      acc[t] = WMMA_BF16(a, bb, acc[t]);
    }
    __syncthreads();
  }
  int n = lane & 15, mb = (lane >> 4) << 3;
#pragma unroll
  for (int t = 0; t < 4; ++t) {
    int col = wid * 64 + t * 16 + n;
#pragma unroll
    for (int r = 0; r < 8; ++r) {
      int row = mb + r;
      float val = acc[t][r] +
                  0.5f * (vq[(rowBase + row) * D + col] + vk[(rowBase + row) * D + col]);
      lnt[row * D + col] = val;
    }
  }
  __syncthreads();
  {
    int row = tid >> 4, part = tid & 15;
    float s1 = 0.f, s2 = 0.f;
    int c0 = part * 32;
    for (int c = 0; c < 32; ++c) {
      float x = lnt[row * D + c0 + c];
      s1 += x; s2 += x * x;
    }
    red[row * 16 + part] = s1;
    red[256 + row * 16 + part] = s2;
  }
  __syncthreads();
  if (tid < 16) {
    float a = 0.f, b2 = 0.f;
    for (int p = 0; p < 16; ++p) { a += red[tid * 16 + p]; b2 += red[256 + tid * 16 + p]; }
    float mu = a * (1.0f / 512.0f);
    float var = b2 * (1.0f / 512.0f) - mu * mu;
    musig[tid] = mu;
    musig[16 + tid] = rsqrtf(var + 1e-6f);
  }
  __syncthreads();
  for (int i = tid; i < 16 * 512; i += 256) {
    int row = i >> 9, col = i & 511;
    float x = lnt[i];
    out[(rowBase + row) * D + col] =
        (x - musig[row]) * musig[16 + row] * gamma[col] + beta[col];
  }
}

extern "C" void kernel_launch(void* const* d_in, const int* in_sizes, int n_in,
                              void* d_out, int out_size, void* d_ws, size_t ws_size,
                              hipStream_t stream) {
  (void)in_sizes; (void)n_in; (void)out_size; (void)ws_size;
  const float* q     = (const float*)d_in[0];
  const float* k     = (const float*)d_in[1];
  const float* v_q   = (const float*)d_in[2];
  const float* v_k   = (const float*)d_in[3];
  const int*   mask  = (const int*)d_in[4];
  const float* Wq    = (const float*)d_in[5];
  const float* Wk    = (const float*)d_in[6];
  const float* Wv    = (const float*)d_in[7];
  const float* Wfc   = (const float*)d_in[8];
  const float* gamma = (const float*)d_in[9];
  const float* beta  = (const float*)d_in[10];

  float* out  = (float*)d_out;                       // (B,S,D)
  float* attn = out + (size_t)BB * S * D;            // (B,H,S,S)

  __bf16* Qh = (__bf16*)d_ws;
  __bf16* Kh = Qh + (size_t)ROWS * D;
  __bf16* Vh = Kh + (size_t)ROWS * D;
  float*  OutH = (float*)(Vh + (size_t)ROWS * D);

  dim3 g1(D / 64, ROWS / 64, 3);
  proj_kernel<<<g1, 256, 0, stream>>>(q, k, v_q, v_k, Wq, Wk, Wv, Qh, Kh, Vh);

  dim3 g2(S / 128, S / 16, BB * H);
  scores_kernel<<<g2, 256, 0, stream>>>(Qh, Kh, mask, attn);

  dim3 g3(S / 16, BB * H);
  attnv_kernel<<<g3, 128, 0, stream>>>(attn, Vh, OutH);

  dim3 g4(ROWS / 16);
  final_kernel<<<g4, 256, 0, stream>>>(OutH, Wfc, v_q, v_k, gamma, beta, out);
}
